// LinearAttention_37967510896789
// MI455X (gfx1250) — compile-verified, matrix operand order checked by reading, not run on hardware
//
#include <hip/hip_runtime.h>
#include <hip/hip_bf16.h>

typedef __attribute__((ext_vector_type(16))) __bf16 v16bf;
typedef __attribute__((ext_vector_type(8)))  float  v8f;

#define B_   4
#define N_   8192
#define D_   512
#define H_   8
#define DH_  64
#define TRI  1536
#define R_   (B_ * N_)          // 32768 flattened rows
#define LDT  80                 // LDS row stride (bf16): 160 B = 16-B aligned rows

// 16x32 (K=32) bf16 fragment gather from a K-contiguous LDS row.
// CDNA5 16-bit A layout: lanes 0-15 hold K=0..7,16..23 ; lanes 16-31 hold
// K=8..15,24..31 (packed pairs).  B uses the same gather because B tiles are
// stored N-major (transposed) in LDS.  All groups are 16-B aligned ->
// compiler lowers to ds_load_b128 pairs.
__device__ __forceinline__ v16bf frag_row(const __bf16* row) {
  const int lane = threadIdx.x & 31;
  const int half = (lane >> 4) & 1;
  v16bf f;
#pragma unroll
  for (int v = 0; v < 8; ++v) {
    const int k = ((v < 4) ? (2 * v) : (16 + 2 * (v - 4))) + half * 8;
    f[2 * v]     = row[k];
    f[2 * v + 1] = row[k + 1];
  }
  return f;
}

__device__ __forceinline__ v8f wmma_bf16(v16bf a, v16bf b, v8f c) {
  return __builtin_amdgcn_wmma_f32_16x16x32_bf16(false, a, false, b,
                                                 (short)0, c, false, false);
}

// 8-byte LDS/global copy of 4 bf16
__device__ __forceinline__ void cp8(__bf16* dst, const __bf16* src) {
  *reinterpret_cast<uint2*>(dst) = *reinterpret_cast<const uint2*>(src);
}

// ---------------------------------------------------------------- cvt x
__global__ void __launch_bounds__(256)
cvt_x_kernel(const float* __restrict__ x, __bf16* __restrict__ xb) {
  const size_t i = (size_t)(blockIdx.x * 256 + threadIdx.x) * 4;
  const float4 v = *reinterpret_cast<const float4*>(&x[i]);
  alignas(8) __bf16 t[4] = {(__bf16)v.x, (__bf16)v.y, (__bf16)v.z, (__bf16)v.w};
  *reinterpret_cast<uint2*>(&xb[i]) = *reinterpret_cast<const uint2*>(t);
}

// ------------------------------------------------------- cvt+transpose W
// wqbT[n][k] = wq[k][n]  (N-major so GEMM B-tiles stage as contiguous copies)
__global__ void __launch_bounds__(256)
cvt_w_kernel(const float* __restrict__ wq, const float* __restrict__ wo,
             __bf16* __restrict__ wqbT, __bf16* __restrict__ wobT) {
  const int i = blockIdx.x * 256 + threadIdx.x;
  if (i < D_ * TRI) {
    const int n = i >> 9, k = i & 511;
    wqbT[i] = (__bf16)wq[(size_t)k * TRI + n];
  }
  if (i < D_ * D_) {
    const int n = i >> 9, k = i & 511;
    wobT[i] = (__bf16)wo[(size_t)k * D_ + n];
  }
}

// ---------------------------------------------------------------- kernel 1
// qkv = x @ w_qkv ; elu(q)+1, elu(k)+1 ; scatter to [b,h,n,64] bf16
// 256x64 block tile, 32x64 per wave (8 WMMA per K-step)
__global__ void __launch_bounds__(256)
qkv_kernel(const __bf16* __restrict__ xb, const __bf16* __restrict__ wqbT,
           __bf16* __restrict__ qa, __bf16* __restrict__ ka,
           __bf16* __restrict__ va) {
  __shared__ alignas(16) __bf16 As[256][LDT];
  __shared__ alignas(16) __bf16 Bs[64][LDT];

  const int tid  = threadIdx.x;
  const int wave = tid >> 5;
  const int lane = tid & 31;
  const int half = (lane >> 4) & 1;
  const int nlo  = lane & 15;

  const int rowBase = blockIdx.x * 256;
  const int nbase   = blockIdx.y * 64;

  v8f acc[2][4] = {};

  for (int kb = 0; kb < D_; kb += 64) {
#pragma unroll
    for (int i = tid; i < 4096; i += 256) {        // 256x64 A tile, uint2
      const int r = i >> 4, c4 = (i & 15) * 4;
      cp8(&As[r][c4], &xb[(size_t)(rowBase + r) * D_ + kb + c4]);
    }
#pragma unroll
    for (int i = tid; i < 1024; i += 256) {        // 64x64 B tile, uint2
      const int nn = i >> 4, k4 = (i & 15) * 4;
      cp8(&Bs[nn][k4], &wqbT[(size_t)(nbase + nn) * D_ + kb + k4]);
    }
    if (kb + 64 < D_)                               // global_prefetch_b8
      __builtin_prefetch(&xb[(size_t)(rowBase + tid) * D_ + kb + 64], 0, 1);
    __syncthreads();
#pragma unroll
    for (int ks = 0; ks < 64; ks += 32) {
      v16bf a0 = frag_row(&As[wave * 32 + nlo][ks]);
      v16bf a1 = frag_row(&As[wave * 32 + 16 + nlo][ks]);
#pragma unroll
      for (int s = 0; s < 4; ++s) {
        v16bf b = frag_row(&Bs[s * 16 + nlo][ks]);
        acc[0][s] = wmma_bf16(a0, b, acc[0][s]);
        acc[1][s] = wmma_bf16(a1, b, acc[1][s]);
      }
    }
    __syncthreads();
  }

  const int seg = nbase >> 9;                       // 0=q,1=k,2=v
  const int h   = (nbase & 511) >> 6;
  __bf16* dst = (seg == 0) ? qa : (seg == 1) ? ka : va;
#pragma unroll
  for (int st = 0; st < 2; ++st) {
#pragma unroll
    for (int s = 0; s < 4; ++s) {
#pragma unroll
      for (int r = 0; r < 8; ++r) {
        const int gr = rowBase + wave * 32 + st * 16 + r + half * 8;
        const int b  = gr >> 13;
        const int n  = gr & (N_ - 1);
        float v = acc[st][s][r];
        if (seg < 2) v = (v > 0.0f) ? (v + 1.0f) : __expf(v);  // elu+1
        dst[((size_t)(b * H_ + h) * N_ + n) * DH_ + s * 16 + nlo] = (__bf16)v;
      }
    }
  }
}

// ---------------------------------------------------------------- kernel 2
// kvT[b,h][m][d] = (k^T v)[d][m] (K=8192) ; ksum folded into tile staging
__global__ void __launch_bounds__(256)
kv_kernel(const __bf16* __restrict__ ka, const __bf16* __restrict__ va,
          __bf16* __restrict__ kvT, float* __restrict__ ksum) {
  __shared__ alignas(16) __bf16 KT[64][LDT];   // [d][seq]
  __shared__ alignas(16) __bf16 VT[64][LDT];   // [m][seq]
  __shared__ float red[4][64];

  const int tid  = threadIdx.x;
  const int wave = tid >> 5;
  const int lane = tid & 31;
  const int half = (lane >> 4) & 1;
  const int nlo  = lane & 15;

  const int bh = blockIdx.x;                    // 0..31
  const size_t base = (size_t)bh * N_ * DH_;

  const int t0     = wave * 2;                  // wave owns 2 of 16 tiles
  const int dstrip = t0 >> 2;
  const int m0     = t0 & 3;
  const int m1     = m0 + 1;

  v8f acc0 = {}, acc1 = {};
  float ks_acc = 0.f;                           // thread's d = tid&63 is fixed

  for (int nb = 0; nb < N_; nb += 64) {
#pragma unroll
    for (int i = tid; i < 64 * 64; i += 256) {
      const int sq = i >> 6, d = i & 63;
      const __bf16 kvl = ka[base + (size_t)(nb + sq) * DH_ + d];
      KT[d][sq] = kvl;
      ks_acc += (float)kvl;
      VT[d][sq] = va[base + (size_t)(nb + sq) * DH_ + d];
    }
    __syncthreads();
#pragma unroll
    for (int ks = 0; ks < 64; ks += 32) {
      v16bf a  = frag_row(&KT[dstrip * 16 + nlo][ks]);
      v16bf b0 = frag_row(&VT[m0 * 16 + nlo][ks]);
      v16bf b1 = frag_row(&VT[m1 * 16 + nlo][ks]);
      acc0 = wmma_bf16(a, b0, acc0);
      acc1 = wmma_bf16(a, b1, acc1);
    }
    __syncthreads();
  }

  // store kv transposed: kvT[m][d] so attn stages it contiguously
#pragma unroll
  for (int r = 0; r < 8; ++r) {
    const int d = dstrip * 16 + r + half * 8;
    kvT[((size_t)bh * 64 + m0 * 16 + nlo) * 64 + d] = (__bf16)acc0[r];
    kvT[((size_t)bh * 64 + m1 * 16 + nlo) * 64 + d] = (__bf16)acc1[r];
  }

  red[tid >> 6][tid & 63] = ks_acc;
  __syncthreads();
  if (tid < 64) ksum[bh * 64 + tid] = red[0][tid] + red[1][tid] + red[2][tid] + red[3][tid];
}

// ---------------------------------------------------------------- kernel 3
// o = (q @ kv) * z ,  z = 1/(q . ksum + 1e-6)  -> oa as [r,512] bf16
__global__ void __launch_bounds__(256)
attn_kernel(const __bf16* __restrict__ qa, const __bf16* __restrict__ kvT,
            const float* __restrict__ ksum, __bf16* __restrict__ oa) {
  __shared__ alignas(16) __bf16 Qs[256][LDT];
  __shared__ alignas(16) __bf16 KVT[64][LDT];   // [m][d]
  __shared__ float kss[64];
  __shared__ float zs[256];

  const int tid  = threadIdx.x;
  const int wave = tid >> 5;
  const int lane = tid & 31;
  const int half = (lane >> 4) & 1;
  const int nlo  = lane & 15;

  const int rowBase = blockIdx.x * 256;
  const int h  = blockIdx.y;
  const int b  = rowBase >> 13;
  const int n0 = rowBase & (N_ - 1);
  const int bh = b * H_ + h;
  const size_t qbase = (size_t)bh * N_ * DH_;

#pragma unroll
  for (int i = tid; i < 4096; i += 256) {        // 256x64 Q tile
    const int r = i >> 4, c4 = (i & 15) * 4;
    cp8(&Qs[r][c4], &qa[qbase + (size_t)(n0 + r) * DH_ + c4]);
  }
#pragma unroll
  for (int i = tid; i < 1024; i += 256) {        // 64x64 kv tile
    const int m = i >> 4, d4 = (i & 15) * 4;
    cp8(&KVT[m][d4], &kvT[((size_t)bh * 64 + m) * 64 + d4]);
  }
  if (tid < 64) kss[tid] = ksum[bh * 64 + tid];
  __syncthreads();

  {  // normalizer: one thread per row
    float s = 0.f;
#pragma unroll
    for (int c = 0; c < 64; ++c) s += (float)Qs[tid][c] * kss[c];
    zs[tid] = 1.0f / (s + 1e-6f);
  }
  __syncthreads();

  v8f acc[2][4] = {};
#pragma unroll
  for (int ks = 0; ks < 64; ks += 32) {
    v16bf a0 = frag_row(&Qs[wave * 32 + nlo][ks]);
    v16bf a1 = frag_row(&Qs[wave * 32 + 16 + nlo][ks]);
#pragma unroll
    for (int s = 0; s < 4; ++s) {
      v16bf b = frag_row(&KVT[s * 16 + nlo][ks]);
      acc[0][s] = wmma_bf16(a0, b, acc[0][s]);
      acc[1][s] = wmma_bf16(a1, b, acc[1][s]);
    }
  }

#pragma unroll
  for (int st = 0; st < 2; ++st) {
#pragma unroll
    for (int s = 0; s < 4; ++s) {
#pragma unroll
      for (int r = 0; r < 8; ++r) {
        const int m = wave * 32 + st * 16 + r + half * 8;
        const float o = acc[st][s][r] * zs[m];
        oa[(size_t)(rowBase + m) * D_ + h * DH_ + s * 16 + nlo] = (__bf16)o;
      }
    }
  }
}

// ---------------------------------------------------------------- kernel 4
// out = o @ w_out + b_out  (fp32 output)
__global__ void __launch_bounds__(256)
proj_kernel(const __bf16* __restrict__ oa, const __bf16* __restrict__ wobT,
            const float* __restrict__ bout, float* __restrict__ out) {
  __shared__ alignas(16) __bf16 As[256][LDT];
  __shared__ alignas(16) __bf16 Bs[64][LDT];

  const int tid  = threadIdx.x;
  const int wave = tid >> 5;
  const int lane = tid & 31;
  const int half = (lane >> 4) & 1;
  const int nlo  = lane & 15;

  const int rowBase = blockIdx.x * 256;
  const int nbase   = blockIdx.y * 64;

  v8f acc[2][4] = {};
  for (int kb = 0; kb < D_; kb += 64) {
#pragma unroll
    for (int i = tid; i < 4096; i += 256) {
      const int r = i >> 4, c4 = (i & 15) * 4;
      cp8(&As[r][c4], &oa[(size_t)(rowBase + r) * D_ + kb + c4]);
    }
#pragma unroll
    for (int i = tid; i < 1024; i += 256) {
      const int nn = i >> 4, k4 = (i & 15) * 4;
      cp8(&Bs[nn][k4], &wobT[(size_t)(nbase + nn) * D_ + kb + k4]);
    }
    if (kb + 64 < D_)
      __builtin_prefetch(&oa[(size_t)(rowBase + tid) * D_ + kb + 64], 0, 1);
    __syncthreads();
#pragma unroll
    for (int ks = 0; ks < 64; ks += 32) {
      v16bf a0 = frag_row(&As[wave * 32 + nlo][ks]);
      v16bf a1 = frag_row(&As[wave * 32 + 16 + nlo][ks]);
#pragma unroll
      for (int s = 0; s < 4; ++s) {
        v16bf b = frag_row(&Bs[s * 16 + nlo][ks]);
        acc[0][s] = wmma_bf16(a0, b, acc[0][s]);
        acc[1][s] = wmma_bf16(a1, b, acc[1][s]);
      }
    }
    __syncthreads();
  }

#pragma unroll
  for (int st = 0; st < 2; ++st) {
#pragma unroll
    for (int s = 0; s < 4; ++s) {
#pragma unroll
      for (int r = 0; r < 8; ++r) {
        const int gr  = rowBase + wave * 32 + st * 16 + r + half * 8;
        const int col = nbase + s * 16 + nlo;
        out[(size_t)gr * D_ + col] = acc[st][s][r] + bout[col];
      }
    }
  }
}

// ---------------------------------------------------------------- launch
extern "C" void kernel_launch(void* const* d_in, const int* in_sizes, int n_in,
                              void* d_out, int out_size, void* d_ws, size_t ws_size,
                              hipStream_t stream) {
  (void)in_sizes; (void)n_in; (void)out_size; (void)ws_size;
  const float* x     = (const float*)d_in[0];
  const float* w_qkv = (const float*)d_in[1];
  const float* w_out = (const float*)d_in[2];
  const float* b_out = (const float*)d_in[3];
  float* out = (float*)d_out;

  __bf16* ws = (__bf16*)d_ws;
  const size_t BIG = (size_t)R_ * D_;          // 16,777,216 bf16 elems
  __bf16* xb   = ws;                           // x as bf16        [r,512]
  __bf16* qa   = ws + 1 * BIG;                 // elu(q)+1         [b,h,n,64]
  __bf16* ka   = ws + 2 * BIG;                 // elu(k)+1         [b,h,n,64]
  __bf16* va   = ws + 3 * BIG;                 // v                [b,h,n,64]
  __bf16* oa   = ws + 4 * BIG;                 // attn out         [r,512]
  __bf16* kvT  = ws + 5 * BIG;                 // kv transposed    [bh,m,d]
  __bf16* wqbT = kvT + (size_t)B_ * H_ * DH_ * DH_;   // [n,k] 1536x512
  __bf16* wobT = wqbT + (size_t)D_ * TRI;             // [n,k] 512x512
  float*  ksum = (float*)(wobT + (size_t)D_ * D_);    // [bh,64]

  cvt_x_kernel<<<dim3(R_ * D_ / 4 / 256), dim3(256), 0, stream>>>(x, xb);
  cvt_w_kernel<<<dim3((D_ * TRI + 255) / 256), dim3(256), 0, stream>>>(
      w_qkv, w_out, wqbT, wobT);
  qkv_kernel<<<dim3(R_ / 256, TRI / 64), dim3(256), 0, stream>>>(
      xb, wqbT, qa, ka, va);
  kv_kernel<<<dim3(B_ * H_), dim3(256), 0, stream>>>(ka, va, kvT, ksum);
  attn_kernel<<<dim3(R_ / 256, H_), dim3(256), 0, stream>>>(qa, kvT, ksum, oa);
  proj_kernel<<<dim3(R_ / 256, D_ / 64), dim3(256), 0, stream>>>(
      oa, wobT, b_out, out);
}
